// Model_10299331575989
// MI455X (gfx1250) — matrix operand order repaired and checked by reading, not compiled, not run on hardware
//
#include <hip/hip_runtime.h>
#include <hip/hip_bf16.h>
#include <math.h>

// ---------------------------------------------------------------------------
// Longformer sliding-window attention layer, fully fused for MI455X (gfx1250).
// B=16, S=8192, D=192, H=16, HD=12, window W=2 (5 diagonals).
//
//  * bf16 WMMA (v_wmma_f32_16x16x32_bf16) for all four 192x192 GEMMs,
//    two N-tiles per wave iteration (shared A frag, 2 independent acc chains)
//  * Tensor Data Mover (tensor_load_to_lds, 6-arg clang-23 form) stages the
//    bf16 weights from L2 into LDS, tracked with TENSORcnt
//  * one workgroup = 64 sequence rows of one batch, K/V recomputed with an
//    8-row halo so q/k/v never touch HBM
//  * banded scores/softmax/PV in VALU from LDS; output projection + residual
//    + LayerNorm fused in the same kernel
// ---------------------------------------------------------------------------

#define BATCH   16
#define SEQ     8192
#define DMODEL  192
#define NHEAD   16
#define HDIM    12
#define ROWS    64
#define HALO    8
#define KVROWS  (ROWS + 2*HALO)       // 80
#define LSTR    200                   // LDS row stride (bf16 elems, 400B)
#define WMAT    (DMODEL * LSTR)       // one padded weight matrix, elements
#define NWAVES  8
#define NTHREADS 256
#define QSCALE  0.2886751345948129f   // 1/sqrt(12)
#define NEGF    (-3.4028234663852886e38f)

typedef __attribute__((ext_vector_type(16))) __bf16 bf16x16;
typedef __attribute__((ext_vector_type(8)))  __bf16 bf16x8;
typedef __attribute__((ext_vector_type(4)))  __bf16 bf16x4;
typedef __attribute__((ext_vector_type(8)))  float  f32x8;

union Frag { bf16x16 v; bf16x8 h[2]; };
union Pack4 { bf16x4 v; __bf16 e[4]; };

// ---- LDS layout (bytes). Total 305,920 <= 320KB --------------------------
#define OFF_XBF   0                    // 80  x 400 = 32000
#define OFF_WA    32000                // 192 x 400 = 76800 (Wq, then Wv)
#define OFF_WB    108800               // 192 x 400 = 76800 (Wk, then Wo)
#define OFF_QS    185600               // 64  x 400 = 25600
#define OFF_KS    211200               // 80  x 400 = 32000
#define OFF_VS    243200               // 80  x 400 = 32000
#define OFF_CTX   275200               // 64  x 400 = 25600
#define OFF_BIAS  300800               // 4 x 192 x 4 = 3072
#define OFF_RED   303872               // 64 x 4 x 2 x 4 = 2048
#define OFF_HBUF  OFF_QS               // fp32 64x192 = 49152 aliases QS+KS
#define SMEM_BYTES 305920

// ---------------------------------------------------------------------------
// Kernel 0: convert weights to bf16 in workspace, already in the padded
// stride-200 layout so TDM can stage them into LDS as flat 1-D tiles.
// ---------------------------------------------------------------------------
__global__ void cvt_weights(const float* __restrict__ Wq,
                            const float* __restrict__ Wk,
                            const float* __restrict__ Wv,
                            const float* __restrict__ Wo,
                            __bf16* __restrict__ out) {
    const int n = DMODEL * DMODEL;
    for (int i = blockIdx.x * blockDim.x + threadIdx.x; i < n;
         i += gridDim.x * blockDim.x) {
        const int r = i / DMODEL, c = i % DMODEL;
        const int o = r * LSTR + c;
        out[o]            = (__bf16)Wq[i];
        out[WMAT + o]     = (__bf16)Wk[i];
        out[2 * WMAT + o] = (__bf16)Wv[i];
        out[3 * WMAT + o] = (__bf16)Wo[i];
    }
}

// ---------------------------------------------------------------------------
// Weight staging: TDM (preferred) or plain VMEM+DS fallback.
// ---------------------------------------------------------------------------
#if __has_builtin(__builtin_amdgcn_tensor_load_to_lds) && \
    __has_builtin(__builtin_amdgcn_s_wait_tensorcnt)
#define HAVE_TDM 1
typedef __attribute__((ext_vector_type(4))) unsigned int u32x4;
typedef __attribute__((ext_vector_type(8))) int i32x8;
typedef __attribute__((ext_vector_type(4))) int i32x4;

// One flat 1-D tile: nelem bf16 elements, global -> LDS byte offset lds_off.
__device__ __forceinline__ void tdm_copy(const __bf16* gsrc,
                                         unsigned lds_off, unsigned nelem) {
    const unsigned long long ga = (unsigned long long)(uintptr_t)gsrc;
    const unsigned lo   = (unsigned)__builtin_amdgcn_readfirstlane((int)lds_off);
    const unsigned galo = (unsigned)__builtin_amdgcn_readfirstlane((int)(ga & 0xffffffffu));
    const unsigned gahi = (unsigned)__builtin_amdgcn_readfirstlane((int)(ga >> 32));
    u32x4 g0;
    g0[0] = 1u;                                   // D# count=1, user mode
    g0[1] = lo;                                   // lds_addr (bytes)
    g0[2] = galo;                                 // global_addr[31:0]
    g0[3] = (gahi & 0x1ffffffu) | (2u << 30);     // global_addr[56:32], type=2
    i32x8 g1;
    g1[0] = (int)(1u << 16);                      // data_size=1 (2B), wg_mask=0
    g1[1] = (int)((nelem & 0xffffu) << 16);       // tensor_dim0[15:0]
    g1[2] = (int)(((nelem >> 16) & 0xffffu) | (1u << 16)); // dim0 hi, tensor_dim1=1
    g1[3] = (int)((nelem & 0xffffu) << 16);       // tile_dim0
    g1[4] = 1;                                    // tile_dim1=1, tile_dim2=0
    g1[5] = (int)nelem;                           // tensor_dim0_stride[31:0]
    g1[6] = 0;
    g1[7] = 0;
    const i32x4 gz4 = {0, 0, 0, 0};
    const i32x8 gz8 = {0, 0, 0, 0, 0, 0, 0, 0};
    __builtin_amdgcn_tensor_load_to_lds(g0, g1, gz4, gz4, gz8, 0);
}

// Stage one 192x200 bf16 matrix: 8 waves each move a 24-row (4800 elem) chunk.
__device__ __forceinline__ void stage_w(const __bf16* gw, unsigned lds_base,
                                        int wid) {
    const unsigned coff = (unsigned)(wid * 24 * LSTR);   // element offset
    tdm_copy(gw + coff, lds_base + coff * 2u, 24u * LSTR);
}
#else
#define HAVE_TDM 0
#endif

// VMEM fallback copy (also used if TDM builtins are absent).
__device__ __forceinline__ void copy_w(const __bf16* __restrict__ gw,
                                       __bf16* lds, int tid) {
    for (int t = tid; t < DMODEL * 25; t += NTHREADS) {  // 25 chunks of 8 bf16
        const int r = t / 25, c = t % 25;
        *(bf16x8*)(lds + r * LSTR + c * 8) =
            *(const bf16x8*)(gw + r * LSTR + c * 8);
    }
}

// ---------------------------------------------------------------------------
// LDS GEMM: D = A(M x 192) @ W^T + bias, operands bf16 in LDS.
// Two N-tiles per iteration: one shared A fragment, two independent
// accumulator chains to hide WMMA->WMMA latency.
// ---------------------------------------------------------------------------
__device__ __forceinline__ void gemm_project(
    const __bf16* A, int aRow0, int mTiles,
    const __bf16* W, const float* bias, float scale,
    __bf16* Dst, int dRow0, int wid, int lane) {
    const int mrow   = lane & 15;
    const int khalfA = (lane >> 4) << 4;     // A frag: K offset 0 or 16
    const int khalfB = (lane >> 4) << 4;     // B frag: K offset 0 or 16
    const int total  = mTiles * 6;           // 6 N-tile pairs over D=192
    for (int t = wid; t < total; t += NWAVES) {
        const int mt = t / 6, np = t % 6;
        const __bf16* aRow  = A + (aRow0 + mt * 16 + mrow) * LSTR + khalfA;
        const __bf16* bRow0 = W + (np * 32 + mrow) * LSTR + khalfB;
        const __bf16* bRow1 = bRow0 + 16 * LSTR;
        f32x8 acc0 = {}, acc1 = {};
        #pragma unroll
        for (int kc = 0; kc < 6; ++kc) {     // K = 192 = 6 x 32
            Frag a, b0, b1;
            a.h[0]  = *(const bf16x8*)(aRow + kc * 32);
            a.h[1]  = *(const bf16x8*)(aRow + kc * 32 + 8);
            b0.h[0] = *(const bf16x8*)(bRow0 + kc * 32);
            b0.h[1] = *(const bf16x8*)(bRow0 + kc * 32 + 8);
            b1.h[0] = *(const bf16x8*)(bRow1 + kc * 32);
            b1.h[1] = *(const bf16x8*)(bRow1 + kc * 32 + 8);
            acc0 = __builtin_amdgcn_wmma_f32_16x16x32_bf16(
                false, a.v, false, b0.v, (short)0, acc0, false, false);
            acc1 = __builtin_amdgcn_wmma_f32_16x16x32_bf16(
                false, a.v, false, b1.v, (short)0, acc1, false, false);
        }
        const int col0 = np * 32 + (lane & 15);
        const int rb   = dRow0 + mt * 16 + ((lane >> 4) << 3);
        const float bb0 = bias[col0], bb1 = bias[col0 + 16];
        #pragma unroll
        for (int r = 0; r < 8; ++r) {
            Dst[(rb + r) * LSTR + col0]      = (__bf16)((acc0[r] + bb0) * scale);
            Dst[(rb + r) * LSTR + col0 + 16] = (__bf16)((acc1[r] + bb1) * scale);
        }
    }
}

// ---------------------------------------------------------------------------
// Kernel 1: fully fused layer.
// ---------------------------------------------------------------------------
__global__ __launch_bounds__(NTHREADS, 1)
void longformer_fused(const float* __restrict__ xg,
                      const float* __restrict__ mask0,
                      const __bf16* __restrict__ wbf,   // [Wq|Wk|Wv|Wo] padded bf16
                      const float* __restrict__ bq, const float* __restrict__ bk,
                      const float* __restrict__ bv, const float* __restrict__ bo,
                      const float* __restrict__ ln_g, const float* __restrict__ ln_b,
                      float* __restrict__ out) {
    __shared__ __align__(16) unsigned char smem[SMEM_BYTES];
    __bf16* XBF = (__bf16*)(smem + OFF_XBF);
    __bf16* WA  = (__bf16*)(smem + OFF_WA);
    __bf16* WB  = (__bf16*)(smem + OFF_WB);
    __bf16* QS  = (__bf16*)(smem + OFF_QS);
    __bf16* KS  = (__bf16*)(smem + OFF_KS);
    __bf16* VS  = (__bf16*)(smem + OFF_VS);
    __bf16* CTX = (__bf16*)(smem + OFF_CTX);
    float*  BIAS = (float*)(smem + OFF_BIAS);
    float*  RED  = (float*)(smem + OFF_RED);
    float*  HBUF = (float*)(smem + OFF_HBUF);

    const int tid  = threadIdx.x;
    const int lane = tid & 31;                    // wave32
    const int wid  = tid >> 5;
    const int bidx = blockIdx.x >> 7;             // 128 row-blocks per batch
    const int r0   = (blockIdx.x & 127) * ROWS;
    const size_t bS = (size_t)bidx * SEQ;

    // ---- Phase 0: stage x tile (halo, bf16), Wq/Wk via TDM, biases --------
#if HAVE_TDM
    stage_w(wbf,        (unsigned)OFF_WA, wid);    // Wq -> WA  (async DMA)
    stage_w(wbf + WMAT, (unsigned)OFF_WB, wid);    // Wk -> WB  (async DMA)
#endif
    for (int i = tid; i < DMODEL; i += NTHREADS) {
        BIAS[i]              = bq[i];
        BIAS[DMODEL + i]     = bk[i];
        BIAS[2 * DMODEL + i] = bv[i];
        BIAS[3 * DMODEL + i] = bo[i];
    }
    for (int t = tid; t < KVROWS * 48; t += NTHREADS) {   // 48 float4 per row
        const int lr = t / 48, cc = t % 48;
        const int gr = r0 - HALO + lr;
        float4 xv = make_float4(0.f, 0.f, 0.f, 0.f);
        if (gr >= 0 && gr < SEQ)
            xv = *(const float4*)(xg + (bS + gr) * DMODEL + cc * 4);
        Pack4 p;
        p.e[0] = (__bf16)xv.x; p.e[1] = (__bf16)xv.y;
        p.e[2] = (__bf16)xv.z; p.e[3] = (__bf16)xv.w;
        *(bf16x4*)(XBF + lr * LSTR + cc * 4) = p.v;
    }
#if HAVE_TDM
    __builtin_amdgcn_s_wait_tensorcnt((short)0);
#else
    copy_w(wbf,        WA, tid);
    copy_w(wbf + WMAT, WB, tid);
#endif
    __syncthreads();

    // ---- Phase 1: Q (interior 64 rows, pre-scaled) and K (80 rows) --------
    gemm_project(XBF, HALO, 4, WA, BIAS,          QSCALE, QS, 0, wid, lane);
    gemm_project(XBF, 0,    5, WB, BIAS + DMODEL, 1.0f,   KS, 0, wid, lane);
    __syncthreads();

    // ---- Phase 2: swap in Wv/Wo (TDM), compute V (80 rows) ----------------
#if HAVE_TDM
    stage_w(wbf + 2 * WMAT, (unsigned)OFF_WA, wid);   // Wv
    stage_w(wbf + 3 * WMAT, (unsigned)OFF_WB, wid);   // Wo
    __builtin_amdgcn_s_wait_tensorcnt((short)0);
#else
    copy_w(wbf + 2 * WMAT, WA, tid);
    copy_w(wbf + 3 * WMAT, WB, tid);
#endif
    __syncthreads();
    gemm_project(XBF, 0, 5, WA, BIAS + 2 * DMODEL, 1.0f, VS, 0, wid, lane);
    __syncthreads();

    // ---- Phase 3: banded scores -> softmax(5) -> PV (per row x head) ------
    for (int t = tid; t < ROWS * NHEAD; t += NTHREADS) {
        const int lr = t >> 4;                    // interior row 0..63
        const int hh = t & 15;
        const int lk = lr + HALO;                 // row in KS/VS
        const int gi = r0 + lr;
        const int c0 = hh * HDIM;
        float qv[HDIM];
        #pragma unroll
        for (int j = 0; j < HDIM; ++j) qv[j] = (float)QS[lr * LSTR + c0 + j];

        float s[5];
        #pragma unroll
        for (int d = 0; d < 5; ++d) {
            const int nk = gi + d - 2;
            if (nk >= 0 && nk < SEQ) {
                const int kr = lk + d - 2;
                float acc = 0.f;
                #pragma unroll
                for (int j = 0; j < HDIM; ++j)
                    acc += qv[j] * (float)KS[kr * LSTR + c0 + j];
                const float mv = mask0[bS + nk];
                s[d] = acc + ((mv != 0.f) ? NEGF : 0.f);
            } else {
                s[d] = -INFINITY;
            }
        }
        float m = s[0];
        #pragma unroll
        for (int d = 1; d < 5; ++d) m = fmaxf(m, s[d]);
        float e[5], sum = 0.f;
        #pragma unroll
        for (int d = 0; d < 5; ++d) { e[d] = __expf(s[d] - m); sum += e[d]; }
        const float keep = (mask0[bS + gi] < 0.f) ? 0.f : 1.f;
        const float inv = keep / sum;

        float cx[HDIM];
        #pragma unroll
        for (int j = 0; j < HDIM; ++j) cx[j] = 0.f;
        #pragma unroll
        for (int d = 0; d < 5; ++d) {
            const float p = e[d] * inv;
            const int vr = lk + d - 2;
            #pragma unroll
            for (int j = 0; j < HDIM; ++j)
                cx[j] += p * (float)VS[vr * LSTR + c0 + j];
        }
        #pragma unroll
        for (int j = 0; j < HDIM; ++j)
            CTX[lr * LSTR + c0 + j] = (__bf16)cx[j];
    }
    __syncthreads();   // ctx ready; QS/KS dead -> HBUF may alias them

    // ---- Phase 4: output projection + bias + residual -> HBUF (fp32) ------
    {
        const int mrow   = lane & 15;
        const int khalfA = (lane >> 4) << 4;
        const int khalfB = (lane >> 4) << 4;
        for (int t = wid; t < 24; t += NWAVES) {      // 4 mt x 6 N-pairs
            const int mt = t / 6, np = t % 6;
            const __bf16* aRow  = CTX + (mt * 16 + mrow) * LSTR + khalfA;
            const __bf16* bRow0 = WB + (np * 32 + mrow) * LSTR + khalfB;
            const __bf16* bRow1 = bRow0 + 16 * LSTR;
            f32x8 acc0 = {}, acc1 = {};
            #pragma unroll
            for (int kc = 0; kc < 6; ++kc) {
                Frag a, b0, b1;
                a.h[0]  = *(const bf16x8*)(aRow + kc * 32);
                a.h[1]  = *(const bf16x8*)(aRow + kc * 32 + 8);
                b0.h[0] = *(const bf16x8*)(bRow0 + kc * 32);
                b0.h[1] = *(const bf16x8*)(bRow0 + kc * 32 + 8);
                b1.h[0] = *(const bf16x8*)(bRow1 + kc * 32);
                b1.h[1] = *(const bf16x8*)(bRow1 + kc * 32 + 8);
                acc0 = __builtin_amdgcn_wmma_f32_16x16x32_bf16(
                    false, a.v, false, b0.v, (short)0, acc0, false, false);
                acc1 = __builtin_amdgcn_wmma_f32_16x16x32_bf16(
                    false, a.v, false, b1.v, (short)0, acc1, false, false);
            }
            const int col0 = np * 32 + (lane & 15);
            const int rb   = mt * 16 + ((lane >> 4) << 3);
            const float bb0 = BIAS[3 * DMODEL + col0];
            const float bb1 = BIAS[3 * DMODEL + col0 + 16];
            #pragma unroll
            for (int r = 0; r < 8; ++r) {
                const int row = rb + r;
                const size_t xo = (bS + r0 + row) * DMODEL;
                HBUF[row * DMODEL + col0]      = acc0[r] + bb0 + xg[xo + col0];
                HBUF[row * DMODEL + col0 + 16] = acc1[r] + bb1 + xg[xo + col0 + 16];
            }
        }
    }
    __syncthreads();

    // ---- Phase 5: LayerNorm over D=192, write output ----------------------
    {
        const int row  = tid >> 2;                // 4 threads per row
        const int part = tid & 3;                 // 48 cols each
        float s1 = 0.f, s2 = 0.f;
        const int cbase = part * 48;
        for (int c = cbase; c < cbase + 48; ++c) {
            const float hv = HBUF[row * DMODEL + c];
            s1 += hv; s2 += hv * hv;
        }
        RED[(row * 4 + part) * 2 + 0] = s1;
        RED[(row * 4 + part) * 2 + 1] = s2;
        __syncthreads();
        float S1 = 0.f, S2 = 0.f;
        #pragma unroll
        for (int p = 0; p < 4; ++p) {
            S1 += RED[(row * 4 + p) * 2 + 0];
            S2 += RED[(row * 4 + p) * 2 + 1];
        }
        const float mu  = S1 * (1.f / DMODEL);
        const float var = S2 * (1.f / DMODEL) - mu * mu;
        const float rs  = rsqrtf(var + 1e-12f);
        const size_t ob = (bS + r0 + row) * DMODEL;
        for (int c = cbase; c < cbase + 48; ++c) {
            const float hv = HBUF[row * DMODEL + c];
            out[ob + c] = (hv - mu) * rs * ln_g[c] + ln_b[c];
        }
    }
}

// ---------------------------------------------------------------------------
extern "C" void kernel_launch(void* const* d_in, const int* in_sizes, int n_in,
                              void* d_out, int out_size, void* d_ws, size_t ws_size,
                              hipStream_t stream) {
    const float* x     = (const float*)d_in[0];
    const float* mask0 = (const float*)d_in[1];
    const float* Wq    = (const float*)d_in[2];
    const float* bq    = (const float*)d_in[3];
    const float* Wk    = (const float*)d_in[4];
    const float* bk    = (const float*)d_in[5];
    const float* Wv    = (const float*)d_in[6];
    const float* bv    = (const float*)d_in[7];
    const float* Wo    = (const float*)d_in[8];
    const float* bo    = (const float*)d_in[9];
    const float* ln_g  = (const float*)d_in[10];
    const float* ln_b  = (const float*)d_in[11];
    float* out = (float*)d_out;
    __bf16* wbf = (__bf16*)d_ws;   // 4 * 192*200 bf16 = 307,200 B

    cvt_weights<<<144, 256, 0, stream>>>(Wq, Wk, Wv, Wo, wbf);

    const int blocks = BATCH * (SEQ / ROWS);     // 2048
    longformer_fused<<<blocks, NTHREADS, 0, stream>>>(
        x, mask0, wbf, bq, bk, bv, bo, ln_g, ln_b, out);
}
